// Embedding2Score_35914516529747
// MI455X (gfx1250) — matrix operand order repaired and checked by reading, not compiled, not run on hardware
//
#include <hip/hip_runtime.h>

typedef __attribute__((ext_vector_type(16))) __bf16 v16bf;
typedef __attribute__((ext_vector_type(8)))  float  v8f;

#define H 128
#define MT 2          // M-tiles (of 16 rows) per k2 block

union ABf {
    unsigned int u[8];
    v16bf v;
};

// round-to-nearest-even f32 -> bf16, packed pair into one dword
__device__ __forceinline__ unsigned int pk_bf16(float x, float y) {
    unsigned int ax = __float_as_uint(x);
    unsigned int ay = __float_as_uint(y);
    ax = (ax + 0x7fffu + ((ax >> 16) & 1u)) >> 16;
    ay = (ay + 0x7fffu + ((ay >> 16) & 1u)) & 0xffff0000u;
    return ax | ay;
}

// ---------------- k0: one-time f32 -> bf16 weight conversion into ws ----------
__global__ void k0_convert_weights(const float* __restrict__ W1,
                                   const float* __restrict__ W2,
                                   const float* __restrict__ W3,
                                   unsigned int* __restrict__ w1bf,
                                   unsigned int* __restrict__ w2bf,
                                   unsigned int* __restrict__ w3bf) {
    int tid = blockIdx.x * blockDim.x + threadIdx.x;
    if (tid < 8192) {                               // W1: 128*128 = 16384 elems
        w1bf[tid] = pk_bf16(W1[2 * tid], W1[2 * tid + 1]);
    } else if (tid < 16384) {                       // W2: 16384 elems
        int i = tid - 8192;
        w2bf[i] = pk_bf16(W2[2 * i], W2[2 * i + 1]);
    } else if (tid < 32768) {                       // W3: 128*256 = 32768 elems
        int i = tid - 16384;
        w3bf[i] = pk_bf16(W3[2 * i], W3[2 * i + 1]);
    }
}

// Load a 16x32 bf16 A K-tile for this lane from a row-major f32 row.
// A layout (contiguous, matches B): lo lanes elems 0..15 = K0..15;
//               hi lanes elems 0..15 = K16..31.
__device__ __forceinline__ void load_A_ktile(ABf& A, const float* rowbase,
                                             int kt, int hi) {
    const float* s0 = rowbase + kt * 32 + hi * 16;
    const float* s1 = s0 + 8;
#pragma unroll
    for (int e = 0; e < 4; ++e) {
        A.u[e]     = pk_bf16(s0[2 * e], s0[2 * e + 1]);
        A.u[e + 4] = pk_bf16(s1[2 * e], s1[2 * e + 1]);
    }
}

// ---------------- k1: avec[g] = v_n @ W1^T + W1_b + W2_b ; zero s_g accum ----
__global__ __launch_bounds__(32)
void k1_gate_bias(const float* __restrict__ node_emb,
                  const int* __restrict__ last_idx,
                  const unsigned short* __restrict__ w1bf,
                  const float* __restrict__ W1_b,
                  const float* __restrict__ W2_b,
                  float* __restrict__ avec,
                  float* __restrict__ out,
                  int Gtot) {
    const int l  = threadIdx.x;
    const int lm = l & 15;
    const int hi = l >> 4;
    const int g0 = blockIdx.x * 16;

    int grow = g0 + lm;
    if (grow > Gtot - 1) grow = Gtot - 1;
    const int li = last_idx[grow];

    ABf A[4];
#pragma unroll
    for (int kt = 0; kt < 4; ++kt)
        load_A_ktile(A[kt], node_emb + (size_t)li * H, kt, hi);

    for (int n = 0; n < 8; ++n) {
        const int c = n * 16 + lm;
        v8f acc = {};
#pragma unroll
        for (int kt = 0; kt < 4; ++kt) {
            // B layout: lanes 0-15 hold K 0..15, lanes 16-31 hold K 16..31
            v16bf B = *(const v16bf*)(w1bf + (size_t)c * H + kt * 32 + hi * 16);
            acc = __builtin_amdgcn_wmma_f32_16x16x32_bf16(
                false, A[kt].v, false, B, (short)0, acc, false, false);
        }
        const float bias = W1_b[c] + W2_b[c];
#pragma unroll
        for (int j = 0; j < 8; ++j) {
            int g = g0 + j + 8 * hi;
            if (g < Gtot) avec[(size_t)g * H + c] = acc[j] + bias;
        }
    }

    // zero the s_g accumulator rows (d_out) for these 16 graphs
    for (int i = l; i < 16 * H; i += 32) {
        size_t idx = (size_t)g0 * H + i;
        if (idx < (size_t)Gtot * H) out[idx] = 0.0f;
    }
}

// ---------------- k2: per-node gate/alpha + segmented scatter-add of s_g -----
// 32 rows (2 M-tiles) per wave: each W2 B-strip feeds 2 WMMAs (halves L2 B traffic)
__global__ __launch_bounds__(32)
void k2_alpha_scatter(const float* __restrict__ node_emb,
                      const float* __restrict__ num_count,
                      const int* __restrict__ segment_ids,
                      const float* __restrict__ q_w,
                      const float* __restrict__ q_bp,
                      const float* __restrict__ avec,
                      const unsigned short* __restrict__ w2bf,
                      float* __restrict__ out,
                      int total) {
    const int l    = threadIdx.x;
    const int lm   = l & 15;
    const int hi   = l >> 4;
    const int row0 = blockIdx.x * (16 * MT);

    int   myseg[MT];
    float mync[MT];
    ABf   A[MT][4];
#pragma unroll
    for (int m = 0; m < MT; ++m) {
        int r = row0 + m * 16 + lm;                  // lanes 16-31 mirror 0-15
        bool valid = (r < total);
        if (r > total - 1) r = total - 1;
        myseg[m] = segment_ids[r];
        mync[m]  = valid ? num_count[r] : 0.0f;
#pragma unroll
        for (int kt = 0; kt < 4; ++kt)
            load_A_ktile(A[m][kt], node_emb + (size_t)r * H, kt, hi);
    }

    // wave-uniform fast path: segment ids are sorted, so first==last => all equal
    const int  seg_first = __shfl(myseg[0], 0, 32);
    const int  seg_last  = __shfl(myseg[MT - 1], 15, 32);
    const bool uniform   = (seg_first == seg_last);

    int srows[MT][8];
#pragma unroll
    for (int m = 0; m < MT; ++m)
#pragma unroll
        for (int j = 0; j < 8; ++j)
            srows[m][j] = __shfl(myseg[m], j + 8 * hi, 32);

    float ap[MT][8];
#pragma unroll
    for (int m = 0; m < MT; ++m)
#pragma unroll
        for (int j = 0; j < 8; ++j) ap[m][j] = 0.0f;

    for (int n = 0; n < 8; ++n) {
        const int c = n * 16 + lm;                   // column this lane owns in C
        v16bf B[4];
#pragma unroll
        for (int kt = 0; kt < 4; ++kt)
            B[kt] = *(const v16bf*)(w2bf + (size_t)c * H + kt * 32 + hi * 16);

        const float qv = q_w[c];
        float av_u = 0.0f;
        if (uniform) av_u = avec[(size_t)seg_first * H + c];

#pragma unroll
        for (int m = 0; m < MT; ++m) {
            v8f acc = {};
#pragma unroll
            for (int kt = 0; kt < 4; ++kt)
                acc = __builtin_amdgcn_wmma_f32_16x16x32_bf16(
                    false, A[m][kt].v, false, B[kt], (short)0, acc, false, false);

            if (uniform) {
#pragma unroll
                for (int j = 0; j < 8; ++j) {
                    float x    = acc[j] + av_u;
                    float gate = 1.0f / (1.0f + __expf(-x));
                    ap[m][j] += gate * qv;
                }
            } else {
#pragma unroll
                for (int j = 0; j < 8; ++j) {
                    float x    = acc[j] + avec[(size_t)srows[m][j] * H + c];
                    float gate = 1.0f / (1.0f + __expf(-x));
                    ap[m][j] += gate * qv;
                }
            }
        }
    }

    // reduce alpha across the 16 lanes of each half; add q_b
    const float qb = q_bp[0];
#pragma unroll
    for (int m = 0; m < MT; ++m)
#pragma unroll
        for (int j = 0; j < 8; ++j) {
            float v = ap[m][j];
            v += __shfl_xor(v, 1, 32);
            v += __shfl_xor(v, 2, 32);
            v += __shfl_xor(v, 4, 32);
            v += __shfl_xor(v, 8, 32);
            ap[m][j] = v + qb;                       // alpha for row (m*16 + j + 8*hi)
        }

    // segmented accumulation: this lane owns columns 4l..4l+3
    float acc4[4] = {0.0f, 0.0f, 0.0f, 0.0f};
    int curseg = seg_first;
#pragma unroll
    for (int m = 0; m < MT; ++m) {
#pragma unroll
        for (int r = 0; r < 16; ++r) {
            int rseg = __shfl(myseg[m], r, 32);      // uniform across lanes
            if (rseg != curseg) {
#pragma unroll
                for (int cc = 0; cc < 4; ++cc) {
                    __hip_atomic_fetch_add(&out[(size_t)curseg * H + 4 * l + cc],
                                           acc4[cc], __ATOMIC_RELAXED,
                                           __HIP_MEMORY_SCOPE_AGENT);
                    acc4[cc] = 0.0f;
                }
                curseg = rseg;
            }
            float alpha = __shfl(ap[m][r & 7], (r < 8) ? 0 : 16, 32);
            float nc    = __shfl(mync[m], r, 32);
            float cf    = alpha * nc;
            int rr = row0 + m * 16 + r;
            if (rr > total - 1) rr = total - 1;      // cf==0 there anyway
            const float* ne = node_emb + (size_t)rr * H + 4 * l;
#pragma unroll
            for (int cc = 0; cc < 4; ++cc) acc4[cc] += cf * ne[cc];
        }
    }
#pragma unroll
    for (int cc = 0; cc < 4; ++cc)
        __hip_atomic_fetch_add(&out[(size_t)curseg * H + 4 * l + cc],
                               acc4[cc], __ATOMIC_RELAXED,
                               __HIP_MEMORY_SCOPE_AGENT);
}

// ---------------- k3: s_h = [v_n | s_g] @ W3^T + W3_b  (in-place over out) ---
__global__ __launch_bounds__(32)
void k3_final(const float* __restrict__ node_emb,
              const int* __restrict__ last_idx,
              const unsigned short* __restrict__ w3bf,
              const float* __restrict__ W3_b,
              float* __restrict__ out,
              int Gtot) {
    const int l  = threadIdx.x;
    const int lm = l & 15;
    const int hi = l >> 4;
    const int g0 = blockIdx.x * 16;

    int grow = g0 + lm;
    if (grow > Gtot - 1) grow = Gtot - 1;
    const int li = last_idx[grow];

    ABf A[8];
#pragma unroll
    for (int kt = 0; kt < 4; ++kt)                       // K 0..127 : v_n
        load_A_ktile(A[kt], node_emb + (size_t)li * H, kt, hi);
#pragma unroll
    for (int kt = 0; kt < 4; ++kt)                       // K 128..255 : s_g
        load_A_ktile(A[4 + kt], out + (size_t)grow * H, kt, hi);

    for (int n = 0; n < 8; ++n) {
        const int c = n * 16 + lm;
        v8f acc = {};
#pragma unroll
        for (int kt = 0; kt < 8; ++kt) {
            v16bf B = *(const v16bf*)(w3bf + (size_t)c * (2 * H) + kt * 32 + hi * 16);
            acc = __builtin_amdgcn_wmma_f32_16x16x32_bf16(
                false, A[kt].v, false, B, (short)0, acc, false, false);
        }
        const float bias = W3_b[c];
#pragma unroll
        for (int j = 0; j < 8; ++j) {
            int g = g0 + j + 8 * hi;
            if (g < Gtot) out[(size_t)g * H + c] = acc[j] + bias;
        }
    }
}

extern "C" void kernel_launch(void* const* d_in, const int* in_sizes, int n_in,
                              void* d_out, int out_size, void* d_ws, size_t ws_size,
                              hipStream_t stream) {
    const float* node_emb    = (const float*)d_in[0];
    const float* num_count   = (const float*)d_in[1];
    // d_in[2] (sections) unused: fixed tiling + segment_ids cover it
    const int*   segment_ids = (const int*)d_in[3];
    const int*   last_idx    = (const int*)d_in[4];
    // d_in[5..8] unused by the reference math
    const float* W1   = (const float*)d_in[9];
    const float* W1_b = (const float*)d_in[10];
    const float* W2   = (const float*)d_in[11];
    const float* W2_b = (const float*)d_in[12];
    const float* q_w  = (const float*)d_in[13];
    const float* q_b  = (const float*)d_in[14];
    const float* W3   = (const float*)d_in[15];
    const float* W3_b = (const float*)d_in[16];
    float* out = (float*)d_out;

    const int TOTAL = in_sizes[1];     // num_count length
    const int G     = in_sizes[4];     // last_idx length

    // workspace layout: avec[G*128] f32, then bf16 weight copies (~4.3 MB total)
    char* ws = (char*)d_ws;
    float*          avec = (float*)ws;
    unsigned short* w1bf = (unsigned short*)(ws + (size_t)G * H * sizeof(float));
    unsigned short* w2bf = w1bf + 128 * 128;
    unsigned short* w3bf = w2bf + 128 * 128;

    k0_convert_weights<<<128, 256, 0, stream>>>(W1, W2, W3,
                                                (unsigned int*)w1bf,
                                                (unsigned int*)w2bf,
                                                (unsigned int*)w3bf);
    k1_gate_bias<<<(G + 15) / 16, 32, 0, stream>>>(node_emb, last_idx, w1bf,
                                                   W1_b, W2_b, avec, out, G);
    k2_alpha_scatter<<<(TOTAL + 16 * MT - 1) / (16 * MT), 32, 0, stream>>>(
        node_emb, num_count, segment_ids, q_w, q_b, avec, w2bf, out, TOTAL);
    k3_final<<<(G + 15) / 16, 32, 0, stream>>>(node_emb, last_idx, w3bf,
                                               W3_b, out, G);
}
